// SparseAttentionDecoder_37752762532566
// MI455X (gfx1250) — matrix operand order repaired and checked by reading, not compiled, not run on hardware
//
#include <hip/hip_runtime.h>

// ---------------------------------------------------------------------------
// Types / helpers
// ---------------------------------------------------------------------------
typedef __attribute__((ext_vector_type(16))) __bf16 v16bf;
typedef __attribute__((ext_vector_type(8)))  float  v8f;

__device__ __forceinline__ float bf2f(unsigned int hs) {
    unsigned int u = hs << 16;
    float f;
    __builtin_memcpy(&f, &u, 4);
    return f;
}

__device__ __forceinline__ unsigned short f2bf(float f) {
    unsigned int u;
    __builtin_memcpy(&u, &f, 4);
    unsigned int r = u + 0x7fffu + ((u >> 16) & 1u);   // round-to-nearest-even
    return (unsigned short)(r >> 16);
}

__device__ __forceinline__ void unpack8(uint4 u, float* f) {
    f[0] = bf2f(u.x & 0xffffu); f[1] = bf2f(u.x >> 16);
    f[2] = bf2f(u.y & 0xffffu); f[3] = bf2f(u.y >> 16);
    f[4] = bf2f(u.z & 0xffffu); f[5] = bf2f(u.z >> 16);
    f[6] = bf2f(u.w & 0xffffu); f[7] = bf2f(u.w >> 16);
}

// low 32 bits of a generic pointer to LDS == wave-relative LDS byte offset
__device__ __forceinline__ unsigned lds_off(const void* p) {
    return (unsigned)(unsigned long long)(uintptr_t)p;
}

// ---------------------------------------------------------------------------
// Weight convert + transpose:  Wt[n*K + k] = bf16(W[k*N + n])
// ---------------------------------------------------------------------------
__global__ void conv_transpose_bf16_kernel(const float* __restrict__ W,
                                           unsigned short* __restrict__ Wt,
                                           int K, int N) {
    size_t idx = (size_t)blockIdx.x * blockDim.x + threadIdx.x;
    if (idx >= (size_t)K * N) return;
    int n = (int)(idx % N);
    int k = (int)(idx / N);
    Wt[(size_t)n * K + k] = f2bf(W[idx]);
}

// ---------------------------------------------------------------------------
// Embedding:  h[b,t,:] = embed[x[b,t]] + pos_embed[t]
// ---------------------------------------------------------------------------
__global__ void embed_add_kernel(const int* __restrict__ x,
                                 const float* __restrict__ emb,
                                 const float* __restrict__ pos,
                                 float* __restrict__ h,
                                 int T, int D, size_t total) {
    size_t idx = (size_t)blockIdx.x * blockDim.x + threadIdx.x;
    if (idx >= total) return;
    int d  = (int)(idx % D);
    int bt = (int)(idx / D);
    int t  = bt % T;
    int tok = x[bt];
    h[idx] = emb[(size_t)tok * D + d] + pos[(size_t)t * D + d];
}

// ---------------------------------------------------------------------------
// LayerNorm (one block per row, D = 1024) -> bf16 output
// ---------------------------------------------------------------------------
__global__ __launch_bounds__(256) void layernorm_bf16_kernel(
    const float* __restrict__ X, const float* __restrict__ g,
    const float* __restrict__ be, unsigned short* __restrict__ Y, int D) {
    int row = blockIdx.x;
    const float* xr = X + (size_t)row * D;
    __shared__ float red[256];
    int t = threadIdx.x;

    float s = 0.0f;
    for (int j = t; j < D; j += 256) s += xr[j];
    red[t] = s;
    __syncthreads();
    for (int st = 128; st > 0; st >>= 1) {
        if (t < st) red[t] += red[t + st];
        __syncthreads();
    }
    float mu = red[0] / (float)D;
    __syncthreads();

    float v = 0.0f;
    for (int j = t; j < D; j += 256) { float dd = xr[j] - mu; v += dd * dd; }
    red[t] = v;
    __syncthreads();
    for (int st = 128; st > 0; st >>= 1) {
        if (t < st) red[t] += red[t + st];
        __syncthreads();
    }
    float rstd = rsqrtf(red[0] / (float)D + 1e-5f);

    unsigned short* yr = Y + (size_t)row * D;
    for (int j = t; j < D; j += 256)
        yr[j] = f2bf((xr[j] - mu) * rstd * g[j] + be[j]);
}

// ---------------------------------------------------------------------------
// Tiled bf16 GEMM via v_wmma_f32_16x16x32_bf16 with async global->LDS
// double-buffered tile prefetch (GLOBAL_LOAD_ASYNC_TO_LDS_B128 / ASYNCcnt).
//   C[M,N] = A[M,K](bf16) * Bt[N,K](bf16)^T  (+bias) (GELU?) (+resid)
// Tile 128x128x32, 256 threads = 8 waves, each wave: 2x4 16x16 fragments.
// All dims divide tiles (M=4096, N in {256,1024,4096}, K in {1024,4096}).
// ---------------------------------------------------------------------------
#define TILE_M 128
#define TILE_N 128
#define TILE_K 32

__global__ __launch_bounds__(256) void gemm_bf16_wmma_kernel(
    const unsigned short* __restrict__ A,    // M x K
    const unsigned short* __restrict__ Bt,   // N x K (pre-transposed weights)
    const float* __restrict__ bias,          // N (or null)
    const float* __restrict__ resid,         // M x N (or null)
    float* __restrict__ outF,                // M x N f32 (or null)
    unsigned short* __restrict__ outB,       // M x N bf16 (used if outF null)
    int M, int N, int K, int doGelu) {
    __shared__ unsigned short As[2][TILE_M * TILE_K];   // 2 x 8 KB
    __shared__ unsigned short Bs[2][TILE_N * TILE_K];   // 2 x 8 KB

    const int tid  = threadIdx.x;
    const int lane = tid & 31;
    const int wave = tid >> 5;        // 0..7
    const int wm   = wave >> 1;       // 0..3  (32-row slab)
    const int wn   = wave & 1;        // 0..1  (64-col slab)
    const int half = lane >> 4;
    const int l16  = lane & 15;
    const int blockM = blockIdx.y * TILE_M;
    const int blockN = blockIdx.x * TILE_N;

    // ---- per-thread async-fill addressing (two 16B chunks per tile each) ----
    const int c0 = tid, c1 = tid + 256;               // chunk ids 0..511
    const int row0 = c0 >> 2, kc0 = (c0 & 3) * 8;
    const int row1 = c1 >> 2, kc1 = (c1 & 3) * 8;
    const unsigned short* gA0 = A  + (size_t)(blockM + row0) * K + kc0;
    const unsigned short* gA1 = A  + (size_t)(blockM + row1) * K + kc1;
    const unsigned short* gB0 = Bt + (size_t)(blockN + row0) * K + kc0;
    const unsigned short* gB1 = Bt + (size_t)(blockN + row1) * K + kc1;
    const unsigned baseA = lds_off(&As[0][0]);
    const unsigned baseB = lds_off(&Bs[0][0]);
    const unsigned bufBytes = TILE_M * TILE_K * 2;    // 8192
    const unsigned oA0 = (unsigned)(row0 * TILE_K + kc0) * 2;
    const unsigned oA1 = (unsigned)(row1 * TILE_K + kc1) * 2;
    const unsigned oB0 = oA0, oB1 = oA1;

    auto fill_async = [&](int buf, int k0) {
        unsigned bo = (unsigned)buf * bufBytes;
        asm volatile("global_load_async_to_lds_b128 %0, %1, off"
                     :: "v"(baseA + bo + oA0), "v"(gA0 + k0) : "memory");
        asm volatile("global_load_async_to_lds_b128 %0, %1, off"
                     :: "v"(baseA + bo + oA1), "v"(gA1 + k0) : "memory");
        asm volatile("global_load_async_to_lds_b128 %0, %1, off"
                     :: "v"(baseB + bo + oB0), "v"(gB0 + k0) : "memory");
        asm volatile("global_load_async_to_lds_b128 %0, %1, off"
                     :: "v"(baseB + bo + oB1), "v"(gB1 + k0) : "memory");
    };

    v8f acc[2][4];
#pragma unroll
    for (int i = 0; i < 2; ++i)
#pragma unroll
        for (int j = 0; j < 4; ++j)
#pragma unroll
            for (int r = 0; r < 8; ++r) acc[i][j][r] = 0.0f;

    const int nk = K / TILE_K;
    fill_async(0, 0);                                  // prefetch tile 0

    for (int kt = 0; kt < nk; ++kt) {
        const int cur = kt & 1;
        // barrier A: all waves finished reading buffer cur^1 (compute kt-1),
        // safe to overwrite it with the next prefetch.
        __syncthreads();
        if (kt + 1 < nk) {
            fill_async(cur ^ 1, (kt + 1) * TILE_K);
            // 4 old + 4 new in flight; in-order completion => <=4 means the
            // current buffer's loads (this wave's share) have landed in LDS.
            asm volatile("s_wait_asynccnt 0x4" ::: "memory");
        } else {
            asm volatile("s_wait_asynccnt 0x0" ::: "memory");
        }
        // barrier B: every wave's share of the current buffer is in LDS.
        __syncthreads();

        const unsigned short* Ac = As[cur];
        const unsigned short* Bc = Bs[cur];

        // ---- build fragments per ISA VGPR layouts ----
        union { v16bf v; unsigned int u[8]; } afrag[2], bfrag[4];
#pragma unroll
        for (int fm = 0; fm < 2; ++fm) {
            int m = wm * 32 + fm * 16 + l16;
#pragma unroll
            for (int i = 0; i < 8; ++i) {
                int kk = (i < 4 ? 2 * i : 16 + 2 * (i - 4)) + 8 * half;
                afrag[fm].u[i] = *(const unsigned int*)&Ac[m * TILE_K + kk];
            }
        }
#pragma unroll
        for (int fn = 0; fn < 4; ++fn) {
            int n = wn * 64 + fn * 16 + l16;
#pragma unroll
            for (int i = 0; i < 8; ++i) {
                int kk = (i < 4 ? 2 * i : 16 + 2 * (i - 4)) + 8 * half;
                bfrag[fn].u[i] = *(const unsigned int*)&Bc[n * TILE_K + kk];
            }
        }

        // ---- 8 WMMA per wave per k-step ----
#pragma unroll
        for (int fm = 0; fm < 2; ++fm)
#pragma unroll
            for (int fn = 0; fn < 4; ++fn)
                acc[fm][fn] = __builtin_amdgcn_wmma_f32_16x16x32_bf16(
                    false, afrag[fm].v, false, bfrag[fn].v,
                    (short)0, acc[fm][fn], false, false);
    }

    // ---- epilogue: bias / GELU / residual / store ----
#pragma unroll
    for (int fm = 0; fm < 2; ++fm) {
#pragma unroll
        for (int fn = 0; fn < 4; ++fn) {
#pragma unroll
            for (int r = 0; r < 8; ++r) {
                int gm = blockM + wm * 32 + fm * 16 + 8 * half + r;
                int gn = blockN + wn * 64 + fn * 16 + l16;
                float v = acc[fm][fn][r];
                if (bias) v += bias[gn];
                if (doGelu) v = 0.5f * v * (1.0f + erff(v * 0.70710678118654752f));
                size_t o = (size_t)gm * N + gn;
                if (resid) v += resid[o];
                if (outF) outF[o] = v;
                else      outB[o] = f2bf(v);
            }
        }
    }
}

// ---------------------------------------------------------------------------
// Sparse attention: one thread per (b, head, query). <=137 keys per query
// (local window 128 + strided j%256==0). Online softmax, f32 math, bf16 io.
// ---------------------------------------------------------------------------
__device__ __forceinline__ void attend_one(
    int j, const unsigned short* __restrict__ kbase,
    const unsigned short* __restrict__ vbase,
    const float* __restrict__ q, float* __restrict__ acc,
    float& mmax, float& lsum) {
    const unsigned short* kp = kbase + (size_t)j * 1024;
    float s = 0.0f;
#pragma unroll
    for (int c = 0; c < 8; ++c) {
        float t[8];
        unpack8(*(const uint4*)(kp + c * 8), t);
#pragma unroll
        for (int e = 0; e < 8; ++e) s += q[c * 8 + e] * t[e];
    }
    s *= 0.125f;   // 1/sqrt(64)
    float p;
    if (s > mmax) {
        float f = __expf(mmax - s);
        lsum *= f;
#pragma unroll
        for (int dd = 0; dd < 64; ++dd) acc[dd] *= f;
        mmax = s;
        p = 1.0f;
    } else {
        p = __expf(s - mmax);
    }
    lsum += p;
    const unsigned short* vp = vbase + (size_t)j * 1024;
#pragma unroll
    for (int c = 0; c < 8; ++c) {
        float t[8];
        unpack8(*(const uint4*)(vp + c * 8), t);
#pragma unroll
        for (int e = 0; e < 8; ++e) acc[c * 8 + e] += p * t[e];
    }
}

__global__ __launch_bounds__(128) void sparse_attn_kernel(
    const unsigned short* __restrict__ Q,
    const unsigned short* __restrict__ Kb,
    const unsigned short* __restrict__ Vb,
    unsigned short* __restrict__ O) {
    const int T = 2048, D = 1024, HD = 64;
    int idx = blockIdx.x * 128 + threadIdx.x;   // b*16*2048 total
    int b   = idx >> 15;
    int rem = idx & 32767;
    int hh  = rem >> 11;
    int i   = rem & 2047;

    const unsigned short* qp = Q + (size_t)(b * T + i) * D + hh * HD;
    float q[64];
#pragma unroll
    for (int c = 0; c < 8; ++c) unpack8(*(const uint4*)(qp + c * 8), &q[c * 8]);

    float acc[64];
#pragma unroll
    for (int dd = 0; dd < 64; ++dd) acc[dd] = 0.0f;
    float mmax = -3.0e38f, lsum = 0.0f;

    const unsigned short* kbase = Kb + (size_t)b * T * D + hh * HD;
    const unsigned short* vbase = Vb + (size_t)b * T * D + hh * HD;

    int jloc = i - 128;                 // local window: j >= i-128
    if (jloc < 0) jloc = 0;
    for (int j = 0; j < jloc; j += 256)           // strided global tokens
        attend_one(j, kbase, vbase, q, acc, mmax, lsum);
    for (int j = jloc; j <= i; ++j)               // local causal window
        attend_one(j, kbase, vbase, q, acc, mmax, lsum);

    float inv = 1.0f / lsum;
    unsigned short* op = O + (size_t)(b * T + i) * D + hh * HD;
#pragma unroll
    for (int dd = 0; dd < 64; ++dd) op[dd] = f2bf(acc[dd] * inv);
}

// ---------------------------------------------------------------------------
// Launch
// ---------------------------------------------------------------------------
extern "C" void kernel_launch(void* const* d_in, const int* in_sizes, int n_in,
                              void* d_out, int out_size, void* d_ws, size_t ws_size,
                              hipStream_t stream) {
    (void)in_sizes; (void)n_in; (void)out_size; (void)ws_size;

    const int*   x    = (const int*)d_in[0];
    const float* emb  = (const float*)d_in[1];
    const float* pos  = (const float*)d_in[2];
    const float* Wq   = (const float*)d_in[3];  const float* bq  = (const float*)d_in[4];
    const float* Wk   = (const float*)d_in[5];  const float* bk  = (const float*)d_in[6];
    const float* Wv   = (const float*)d_in[7];  const float* bv  = (const float*)d_in[8];
    const float* Wo   = (const float*)d_in[9];  const float* bo  = (const float*)d_in[10];
    const float* g1   = (const float*)d_in[11]; const float* be1 = (const float*)d_in[12];
    const float* W1   = (const float*)d_in[13]; const float* bf1 = (const float*)d_in[14];
    const float* W2   = (const float*)d_in[15]; const float* bf2 = (const float*)d_in[16];
    const float* g2   = (const float*)d_in[17]; const float* be2 = (const float*)d_in[18];
    const float* gf   = (const float*)d_in[19]; const float* bef = (const float*)d_in[20];
    const float* Wh   = (const float*)d_in[21];
    float* out = (float*)d_out;

    const int D = 1024, T = 2048, B = 2, M = B * T, F = 4096, V = 256, L = 2;

    char* ws = (char*)d_ws;
    size_t off = 0;
    auto alloc = [&](size_t bytes) -> void* {
        void* p = ws + off;
        off = (off + bytes + 255) & ~(size_t)255;
        return p;
    };

    unsigned short *wtq[2], *wtk[2], *wtv[2], *wto[2], *wt1[2], *wt2[2];
    for (int l = 0; l < L; ++l) {
        wtq[l] = (unsigned short*)alloc((size_t)D * D * 2);
        wtk[l] = (unsigned short*)alloc((size_t)D * D * 2);
        wtv[l] = (unsigned short*)alloc((size_t)D * D * 2);
        wto[l] = (unsigned short*)alloc((size_t)D * D * 2);
        wt1[l] = (unsigned short*)alloc((size_t)F * D * 2);   // F x D
        wt2[l] = (unsigned short*)alloc((size_t)D * F * 2);   // D x F
    }
    unsigned short* wth  = (unsigned short*)alloc((size_t)V * D * 2);  // V x D
    float*          hbuf = (float*)alloc((size_t)M * D * 4);
    unsigned short* xln  = (unsigned short*)alloc((size_t)M * D * 2);
    unsigned short* qb   = (unsigned short*)alloc((size_t)M * D * 2);
    unsigned short* kb   = (unsigned short*)alloc((size_t)M * D * 2);
    unsigned short* vb   = (unsigned short*)alloc((size_t)M * D * 2);
    unsigned short* atnb = (unsigned short*)alloc((size_t)M * D * 2);
    unsigned short* ffnb = (unsigned short*)alloc((size_t)M * F * 2);

    // ---- one-time weight convert + transpose to bf16 [N][K] ----
    for (int l = 0; l < L; ++l) {
        conv_transpose_bf16_kernel<<<(D * D + 255) / 256, 256, 0, stream>>>(Wq + (size_t)l * D * D, wtq[l], D, D);
        conv_transpose_bf16_kernel<<<(D * D + 255) / 256, 256, 0, stream>>>(Wk + (size_t)l * D * D, wtk[l], D, D);
        conv_transpose_bf16_kernel<<<(D * D + 255) / 256, 256, 0, stream>>>(Wv + (size_t)l * D * D, wtv[l], D, D);
        conv_transpose_bf16_kernel<<<(D * D + 255) / 256, 256, 0, stream>>>(Wo + (size_t)l * D * D, wto[l], D, D);
        conv_transpose_bf16_kernel<<<(D * F + 255) / 256, 256, 0, stream>>>(W1 + (size_t)l * D * F, wt1[l], D, F);
        conv_transpose_bf16_kernel<<<(F * D + 255) / 256, 256, 0, stream>>>(W2 + (size_t)l * F * D, wt2[l], F, D);
    }
    conv_transpose_bf16_kernel<<<(D * V + 255) / 256, 256, 0, stream>>>(Wh, wth, D, V);

    // ---- embedding ----
    embed_add_kernel<<<(M * D + 255) / 256, 256, 0, stream>>>(x, emb, pos, hbuf, T, D, (size_t)M * D);

    auto gemm = [&](const unsigned short* A, const unsigned short* Bt, const float* bias,
                    const float* resid, float* oF, unsigned short* oB, int N, int K, int gelu) {
        dim3 grid(N / TILE_N, M / TILE_M);
        gemm_bf16_wmma_kernel<<<grid, 256, 0, stream>>>(A, Bt, bias, resid, oF, oB, M, N, K, gelu);
    };

    for (int l = 0; l < L; ++l) {
        layernorm_bf16_kernel<<<M, 256, 0, stream>>>(hbuf, g1 + (size_t)l * D, be1 + (size_t)l * D, xln, D);
        gemm(xln, wtq[l], bq + (size_t)l * D, nullptr, nullptr, qb, D, D, 0);
        gemm(xln, wtk[l], bk + (size_t)l * D, nullptr, nullptr, kb, D, D, 0);
        gemm(xln, wtv[l], bv + (size_t)l * D, nullptr, nullptr, vb, D, D, 0);
        sparse_attn_kernel<<<(B * 16 * T) / 128, 128, 0, stream>>>(qb, kb, vb, atnb);
        gemm(atnb, wto[l], bo + (size_t)l * D, hbuf, hbuf, nullptr, D, D, 0);   // h += attn @ Wo + bo
        layernorm_bf16_kernel<<<M, 256, 0, stream>>>(hbuf, g2 + (size_t)l * D, be2 + (size_t)l * D, xln, D);
        gemm(xln, wt1[l], bf1 + (size_t)l * F, nullptr, nullptr, ffnb, F, D, 1); // gelu(h2 @ W1 + b)
        gemm(ffnb, wt2[l], bf2 + (size_t)l * D, hbuf, hbuf, nullptr, D, F, 0);   // h += ffn @ W2 + b
    }
    layernorm_bf16_kernel<<<M, 256, 0, stream>>>(hbuf, gf, bef, xln, D);
    gemm(xln, wth, nullptr, nullptr, out, nullptr, V, D, 0);                      // logits
}
